// BitLinear_43963285242662
// MI455X (gfx1250) — compile-verified, matrix operand order checked by reading, not run on hardware
//
#include <hip/hip_runtime.h>

// ---- CDNA5 (gfx1250, wave32) WMMA GEMM: y = x * w^T, w exact in f16 ----
// Block tile 256x128, 8 wave32 waves in 4x2 grid, each wave 64x64 (4x4 WMMA
// accumulators). Software-pipelined staging: global loads for slab kt+1 are
// issued BEFORE the 16 WMMAs of slab kt, so s_wait_loadcnt lands after the
// matrix block and HBM latency hides under compute.

typedef __attribute__((ext_vector_type(16))) _Float16 v16h;
typedef __attribute__((ext_vector_type(4)))  _Float16 v4h;
typedef __attribute__((ext_vector_type(8)))  float    v8f;
typedef __attribute__((ext_vector_type(4)))  float    v4f;
typedef __attribute__((ext_vector_type(4)))  int      v4i;

// v16h with alignment reduced to 16B: fragment rows in LDS are 16B- but not
// 32B-aligned (row stride 80B); loads lower to 2x ds_load_b128.
typedef v16h v16h_u __attribute__((aligned(16)));

#define BM 256
#define BN 128
#define BK 32
#define LDK 40  // BK + 8 halfs pad -> 80B row stride, bank-conflict-free b128 reads

__global__ __launch_bounds__(256)
void bitlinear_wmma_f16(const float* __restrict__ X,  // [M, K] f32
                        const int*   __restrict__ W,  // [N, K] int32 (values 0..255)
                        float*       __restrict__ Y,  // [M, N] f32
                        int M, int N, int K)
{
  __shared__ _Float16 As[2][BM * LDK];  // 2 x 256 x 40 halfs = 40 KB
  __shared__ _Float16 Bs[2][BN * LDK];  // 2 x 128 x 40 halfs = 20 KB

  const int tid  = threadIdx.x;
  const int lane = tid & 31;
  const int wave = tid >> 5;
  const int wm   = wave >> 1;  // 0..3 -> 64 output rows per wave
  const int wn   = wave & 1;   // 0..1 -> 64 output cols per wave

  const int m0 = blockIdx.x * BM;
  const int n0 = blockIdx.y * BN;

  // Staging: 256 threads, 8 threads x float4 per row -> 32 rows per pass.
  const int srow = tid >> 3;        // 0..31
  const int scol = (tid & 7) << 2;  // 0,4,...,28  (elements)

  v8f acc[4][4];
  {
    v8f z = {0.f, 0.f, 0.f, 0.f, 0.f, 0.f, 0.f, 0.f};
#pragma unroll
    for (int i = 0; i < 4; ++i)
#pragma unroll
      for (int j = 0; j < 4; ++j)
        acc[i][j] = z;
  }

  // In-flight staging registers for one BK-slab (48 VGPRs).
  v4f xr[8];
  v4i wr[4];

  // Issue all 12 global_load_b128 for slab kt into distinct registers.
  auto gload = [&](int kt) {
    const int k0 = kt * BK;
#pragma unroll
    for (int p = 0; p < 8; ++p) {
      const int r = (p << 5) + srow;  // 0..255
      xr[p] = *(const v4f*)(X + (size_t)(m0 + r) * K + (k0 + scol));
    }
#pragma unroll
    for (int p = 0; p < 4; ++p) {
      const int r = (p << 5) + srow;  // 0..127
      wr[p] = *(const v4i*)(W + (size_t)(n0 + r) * K + (k0 + scol));
    }
  };

  // Convert f32/int32 -> f16 (0..255 exact in f16) and store slab to LDS buf.
  auto cvt_store = [&](int buf) {
#pragma unroll
    for (int p = 0; p < 8; ++p) {
      const int r = (p << 5) + srow;
      v4h xh = { (_Float16)xr[p][0], (_Float16)xr[p][1],
                 (_Float16)xr[p][2], (_Float16)xr[p][3] };
      *(v4h*)(&As[buf][r * LDK + scol]) = xh;
    }
#pragma unroll
    for (int p = 0; p < 4; ++p) {
      const int r = (p << 5) + srow;
      v4h wh = { (_Float16)wr[p][0], (_Float16)wr[p][1],
                 (_Float16)wr[p][2], (_Float16)wr[p][3] };
      *(v4h*)(&Bs[buf][r * LDK + scol]) = wh;
    }
  };

  // Fragment addressing, contiguous k order (not the ISA table order), A 16x32:
  // lanes 0-15: rows M=0..15, K=0..15; lanes 16-31: same rows, K=16..31.
  // B fragments (cols of B = rows of w) use the identical pattern.
  const int lr = lane & 15;
  const int kh = (lane >> 4) << 4;  // 0 or 16 halfs

  auto compute = [&](int buf) {
    v16h a[4], b[4];
#pragma unroll
    for (int i = 0; i < 4; ++i)
      a[i] = *(const v16h_u*)&As[buf][(wm * 64 + i * 16 + lr) * LDK + kh];
#pragma unroll
    for (int j = 0; j < 4; ++j)
      b[j] = *(const v16h_u*)&Bs[buf][(wn * 64 + j * 16 + lr) * LDK + kh];
#pragma unroll
    for (int i = 0; i < 4; ++i)
#pragma unroll
      for (int j = 0; j < 4; ++j)
        acc[i][j] = __builtin_amdgcn_wmma_f32_16x16x32_f16(
            /*neg_a=*/false, a[i], /*neg_b=*/false, b[j],
            /*c_mod=*/(short)0, acc[i][j],
            /*reuse_a=*/false, /*reuse_b=*/false);
  };

  const int NK = K / BK;  // 128 k-steps

  // Prologue: fill buffer 0.
  gload(0);
  cvt_store(0);
  __syncthreads();

  for (int kt = 0; kt < NK; ++kt) {
    const int cur = kt & 1;
    if (kt + 1 < NK) gload(kt + 1);      // loads in flight across the WMMAs
    compute(cur);                         // 16 v_wmma, waits only on dscnt
    if (kt + 1 < NK) cvt_store(cur ^ 1);  // consume loads after compute
    __syncthreads();
  }

  // C/D layout: lanes 0-15 -> N=lane, M=vgpr; lanes 16-31 -> N=lane-16, M=vgpr+8.
  const int rb = m0 + wm * 64 + (lane >> 4) * 8;
  const int cb = n0 + wn * 64 + lr;
#pragma unroll
  for (int i = 0; i < 4; ++i)
#pragma unroll
    for (int j = 0; j < 4; ++j)
#pragma unroll
      for (int r = 0; r < 8; ++r)
        Y[(size_t)(rb + i * 16 + r) * N + (cb + j * 16)] = acc[i][j][r];
}

extern "C" void kernel_launch(void* const* d_in, const int* in_sizes, int n_in,
                              void* d_out, int out_size, void* d_ws, size_t ws_size,
                              hipStream_t stream) {
  const float* x = (const float*)d_in[0];  // [4,2048,4096] f32 -> [8192,4096]
  const int*   w = (const int*)d_in[1];    // [4096,4096] int32
  float*       y = (float*)d_out;          // [8192,4096] f32

  const int K = 4096;
  const int M = in_sizes[0] / K;  // 8192
  const int N = in_sizes[1] / K;  // 4096

  dim3 grid(M / BM, N / BN);  // 32 x 32
  bitlinear_wmma_f16<<<grid, 256, 0, stream>>>(x, w, y, M, N, K);
}
